// SpatialLayoutGraphModule_32796370272767
// MI455X (gfx1250) — compile-verified, matrix operand order checked by reading, not run on hardware
//
#include <hip/hip_runtime.h>
#include <hip/hip_bf16.h>
#include <math.h>

typedef __attribute__((ext_vector_type(16))) _Float16 v16h;
typedef __attribute__((ext_vector_type(8)))  _Float16 v8h;
typedef __attribute__((ext_vector_type(8)))  float    v8f;

#define NN 50000
#define EE 250000
#define DD 300
#define HH 4
#define CC 75
#define KP 320    // padded K
#define LDW 320   // padded output cols in packed weights
#define NT (KP / 32)          // K-blocks per GEMM = 10
#define PACKW (NT * LDW * 32) // packed weight elements (halves) = 102400

// ---------- monotone float<->uint encoding for atomic segment-max ----------
static __device__ __forceinline__ unsigned fenc(float f) {
  unsigned u = __float_as_uint(f);
  return (u & 0x80000000u) ? ~u : (u | 0x80000000u);
}
static __device__ __forceinline__ float fdec(unsigned k) {
  unsigned u = (k & 0x80000000u) ? (k & 0x7FFFFFFFu) : ~k;
  return __uint_as_float(u);
}

// ---------- staging / conversion kernels ----------
__global__ void k_conv_x(const float* __restrict__ x, _Float16* __restrict__ xh) {
  int i = blockIdx.x * blockDim.x + threadIdx.x;           // NN*KP threads
  int n = i / KP, c = i % KP;
  xh[i] = (_Float16)((c < DD) ? x[(size_t)n * DD + c] : 0.f);
}

// Pack a [rows<=320, cols<=320] f32 weight (ld=srcld) into WMMA B-fragment order:
// dst[((t*LDW + c)*2 + khalf)*16 + j], first 8 j -> K = t*32+kh*8+j, next 8 -> +16.
__global__ void k_pack_w(const float* __restrict__ src, _Float16* __restrict__ dst,
                         int rows, int cols, int srcld) {
  int o = blockIdx.x * blockDim.x + threadIdx.x;
  if (o >= PACKW) return;
  int j  = o & 15;
  int kh = (o >> 4) & 1;
  int c  = (o >> 5) % LDW;
  int t  = o / (LDW * 32);
  int krow = t * 32 + kh * 8 + ((j < 8) ? j : (j + 8));
  float v = 0.f;
  if (krow < rows && c < cols) v = src[(size_t)krow * srcld + c];
  dst[o] = (_Float16)v;
}

// Same packing for (WobjA - WobjB)
__global__ void k_pack_wdiff(const float* __restrict__ Wobj, _Float16* __restrict__ dst) {
  int o = blockIdx.x * blockDim.x + threadIdx.x;
  if (o >= PACKW) return;
  int j  = o & 15;
  int kh = (o >> 4) & 1;
  int c  = (o >> 5) % LDW;
  int t  = o / (LDW * 32);
  int krow = t * 32 + kh * 8 + ((j < 8) ? j : (j + 8));
  float v = 0.f;
  if (krow < DD && c < DD)
    v = Wobj[(size_t)krow * DD + c] - Wobj[(size_t)(krow + DD) * DD + c];
  dst[o] = (_Float16)v;
}

// nrm = obj_normals @ Wn + bn   (N x 3 @ 3 x 300), padded to f16 [N, 320]
__global__ void k_normal_h(const float* __restrict__ on3, const float* __restrict__ Wn,
                           const float* __restrict__ bn, _Float16* __restrict__ nrmh) {
  int i = blockIdx.x * blockDim.x + threadIdx.x;           // NN*KP threads
  int n = i / KP, c = i % KP;
  float v = 0.f;
  if (c < DD)
    v = bn[c] + on3[n * 3 + 0] * Wn[c] + on3[n * 3 + 1] * Wn[DD + c] + on3[n * 3 + 2] * Wn[2 * DD + c];
  nrmh[i] = (_Float16)v;
}

// ---------- tiny constant-folding kernels ----------
__global__ void k_vocab_fus(const float* __restrict__ vocab, const float* __restrict__ Wfus,
                            float* __restrict__ out7) {
  int i = blockIdx.x * blockDim.x + threadIdx.x;
  if (i >= 7 * DD) return;
  int r = i / DD, d = i % DD;
  const float* wf = (r < 2) ? Wfus : (Wfus + (size_t)DD * DD);
  float s = 0.f;
  for (int k = 0; k < DD; ++k) s += vocab[r * DD + k] * wf[(size_t)k * DD + d];
  out7[i] = s;
}

__global__ void k_w2_bias(const float* __restrict__ Wxy, const float* __restrict__ Wloc,
                          const float* __restrict__ bxy, const float* __restrict__ bobj,
                          const float* __restrict__ bloc,
                          float* __restrict__ W2, float* __restrict__ bias_s) {
  int i = blockIdx.x * blockDim.x + threadIdx.x;
  if (i >= 3 * DD) return;
  int r = i / DD, d = i % DD;
  if (r < 2) {
    float s = 0.f;
    for (int k = 0; k < DD; ++k) s += Wxy[r * DD + k] * Wloc[(size_t)k * DD + d];
    W2[r * DD + d] = s;
  } else {
    float s = bobj[d] + bloc[d];
    for (int k = 0; k < DD; ++k) s += bxy[k] * Wloc[(size_t)k * DD + d];
    bias_s[d] = s;
  }
}

__global__ void k_we_fold(const float* __restrict__ out7, const float* __restrict__ bfus,
                          const float* __restrict__ We,
                          float* __restrict__ vae, float* __restrict__ vbe,
                          float* __restrict__ V5e, float* __restrict__ bfe) {
  int i = blockIdx.x * blockDim.x + threadIdx.x;
  if (i >= 8 * DD) return;
  int r = i / DD, d = i % DD;
  const float* vec = (r < 7) ? (out7 + r * DD) : bfus;
  float s = 0.f;
  for (int k = 0; k < DD; ++k) s += vec[k] * We[(size_t)k * DD + d];
  if (r == 0) vae[d] = s;
  else if (r == 1) vbe[d] = s;
  else if (r < 7) V5e[(r - 2) * DD + d] = s;
  else bfe[d] = s;
}

// ---------- WMMA GEMM: OUT[M, ldo] = A[M,320]f16 @ packedB (+bias)(+addin) ----------
__global__ void gemm_wmma(const _Float16* __restrict__ A, const _Float16* __restrict__ Bp,
                          const float* __restrict__ bias, const float* __restrict__ addin,
                          float* __restrict__ OUT, int outCols, int ldo) {
  const int lane = threadIdx.x & 31;
  const int wave = threadIdx.x >> 5;
  const int r0 = blockIdx.x * 16;                       // M tile (50000 % 16 == 0)
  const int c0 = (blockIdx.y * 8 + wave) * 16;          // N tile per wave
  if (c0 >= LDW) return;                                // uniform per wave
  const int m  = lane & 15;
  const int kh = lane >> 4;                             // K-half select per ISA layout
  const _Float16* arow  = A + (size_t)(r0 + m) * KP + kh * 8;         // 16B aligned
  const _Float16* bbase = Bp + (size_t)(c0 + m) * 32 + kh * 16;       // 32B aligned
  v8f acc = {};
#pragma unroll
  for (int t = 0; t < NT; ++t) {
    v8h alo = *(const v8h*)(arow + t * 32);
    v8h ahi = *(const v8h*)(arow + t * 32 + 16);
    v16h a = __builtin_shufflevector(alo, ahi, 0, 1, 2, 3, 4, 5, 6, 7,
                                     8, 9, 10, 11, 12, 13, 14, 15);
    v16h b = *(const v16h*)(bbase + (size_t)t * (LDW * 32));
    acc = __builtin_amdgcn_wmma_f32_16x16x32_f16(false, a, false, b, (short)0, acc,
                                                 false, false);
  }
  const int n  = lane & 15;
  const int mb = (lane >> 4) * 8;
  const int col = c0 + n;
  if (col < outCols) {
    float bv = bias ? bias[col] : 0.f;
#pragma unroll
    for (int r = 0; r < 8; ++r) {
      int row = r0 + mb + r;
      float v = acc[r] + bv;
      if (addin) v += addin[(size_t)row * ldo + col];
      OUT[(size_t)row * ldo + col] = v;
    }
  }
}

// ---------- edge pass 1: logits->prob, ef, alpha_raw, segment-max ----------
__global__ void edge_pass1(const int* __restrict__ ei, const float* __restrict__ ea,
                           const float* __restrict__ qn, const float* __restrict__ kn,
                           const float* __restrict__ PD, const float* __restrict__ PS,
                           const float* __restrict__ W2, const float* __restrict__ Wcls,
                           const float* __restrict__ bcls,
                           const float* __restrict__ vae, const float* __restrict__ vbe,
                           const float* __restrict__ V5e, const float* __restrict__ bfe,
                           float* __restrict__ alpha, float* __restrict__ eprob,
                           unsigned* __restrict__ menc) {
  const int e = blockIdx.x * (blockDim.x >> 5) + (threadIdx.x >> 5);
  if (e >= EE) return;
  const int lane = threadIdx.x & 31;
  const int src = ei[e];
  const int dst = ei[EE + e];
  const float a0 = ea[e * 4 + 0], a1 = ea[e * 4 + 1];
  const float a2 = ea[e * 4 + 2], a3 = ea[e * 4 + 3];
  const float topg = (a0 > 0.f) ? 1.f : 0.f;
  const float botg = (a1 < 0.f) ? 1.f : 0.f;

  float lg[5] = {0.f, 0.f, 0.f, 0.f, 0.f};
  for (int c = lane; c < DD; c += 32) {
    float s = PD[(size_t)dst * DD + c] + PS[(size_t)src * DD + c] + a2 * W2[c] + a3 * W2[DD + c];
    float t = tanhf(s);
#pragma unroll
    for (int j = 0; j < 5; ++j) lg[j] += t * Wcls[c * 5 + j];
  }
#pragma unroll
  for (int off = 16; off > 0; off >>= 1) {
#pragma unroll
    for (int j = 0; j < 5; ++j) lg[j] += __shfl_xor(lg[j], off, 32);
  }
#pragma unroll
  for (int j = 0; j < 5; ++j) lg[j] += bcls[j];
  float mx = lg[0];
#pragma unroll
  for (int j = 1; j < 5; ++j) mx = fmaxf(mx, lg[j]);
  float prob[5], psum = 0.f;
#pragma unroll
  for (int j = 0; j < 5; ++j) { prob[j] = __expf(lg[j] - mx); psum += prob[j]; }
  float rp = 1.f / psum;
#pragma unroll
  for (int j = 0; j < 5; ++j) prob[j] *= rp;

  float ah[4] = {0.f, 0.f, 0.f, 0.f};
  for (int c = lane; c < DD; c += 32) {
    float ef = topg * vae[c] + botg * vbe[c] + bfe[c];
#pragma unroll
    for (int j = 0; j < 5; ++j) ef += prob[j] * V5e[j * DD + c];
    float qk = qn[(size_t)dst * DD + c] * (kn[(size_t)src * DD + c] + ef);
    int h = c / CC;
#pragma unroll
    for (int hh = 0; hh < HH; ++hh) ah[hh] += (h == hh) ? qk : 0.f;
  }
#pragma unroll
  for (int off = 16; off > 0; off >>= 1) {
#pragma unroll
    for (int hh = 0; hh < HH; ++hh) ah[hh] += __shfl_xor(ah[hh], off, 32);
  }
  const float inv = 0.11547005383792515f; // 1/sqrt(75)
  if (lane == 0) {
#pragma unroll
    for (int h = 0; h < HH; ++h) {
      float a = ah[h] * inv;
      alpha[(size_t)e * HH + h] = a;
      atomicMax(&menc[(size_t)dst * HH + h], fenc(a));
    }
#pragma unroll
    for (int j = 0; j < 5; ++j) eprob[(size_t)e * 8 + j] = prob[j];
    eprob[(size_t)e * 8 + 5] = topg;
    eprob[(size_t)e * 8 + 6] = botg;
  }
}

// ---------- edge pass 2: exp(alpha - m) and segment-sum denominator ----------
__global__ void edge_pass2(const int* __restrict__ ei, float* __restrict__ alpha,
                           const unsigned* __restrict__ menc, float* __restrict__ den) {
  int i = blockIdx.x * blockDim.x + threadIdx.x;
  if (i >= EE * HH) return;
  int e = i >> 2, h = i & 3;
  int dst = ei[EE + e];
  float mm = fdec(menc[(size_t)dst * HH + h]);
  float ex = __expf(alpha[i] - mm);
  alpha[i] = ex;
  atomicAdd(&den[(size_t)dst * HH + h], ex);
}

// ---------- edge pass 3: weighted scatter of (v + ef) into out ----------
__global__ void edge_pass3(const int* __restrict__ ei, const float* __restrict__ alpha,
                           const float* __restrict__ den, const float* __restrict__ vn,
                           const float* __restrict__ eprob,
                           const float* __restrict__ vae, const float* __restrict__ vbe,
                           const float* __restrict__ V5e, const float* __restrict__ bfe,
                           float* __restrict__ out) {
  const int e = blockIdx.x * (blockDim.x >> 5) + (threadIdx.x >> 5);
  if (e >= EE) return;
  const int lane = threadIdx.x & 31;
  const int src = ei[e];
  const int dst = ei[EE + e];
  float w0 = alpha[(size_t)e * HH + 0] / den[(size_t)dst * HH + 0];
  float w1 = alpha[(size_t)e * HH + 1] / den[(size_t)dst * HH + 1];
  float w2 = alpha[(size_t)e * HH + 2] / den[(size_t)dst * HH + 2];
  float w3 = alpha[(size_t)e * HH + 3] / den[(size_t)dst * HH + 3];
  float prob[5];
#pragma unroll
  for (int j = 0; j < 5; ++j) prob[j] = eprob[(size_t)e * 8 + j];
  const float topg = eprob[(size_t)e * 8 + 5];
  const float botg = eprob[(size_t)e * 8 + 6];
  for (int c = lane; c < DD; c += 32) {
    float ef = topg * vae[c] + botg * vbe[c] + bfe[c];
#pragma unroll
    for (int j = 0; j < 5; ++j) ef += prob[j] * V5e[j * DD + c];
    int h = c / CC;
    float w = (h == 0) ? w0 : (h == 1) ? w1 : (h == 2) ? w2 : w3;
    atomicAdd(&out[(size_t)dst * DD + c], (vn[(size_t)src * DD + c] + ef) * w);
  }
}

extern "C" void kernel_launch(void* const* d_in, const int* in_sizes, int n_in,
                              void* d_out, int out_size, void* d_ws, size_t ws_size,
                              hipStream_t stream) {
  (void)in_sizes; (void)n_in; (void)out_size; (void)ws_size;
  const float* x    = (const float*)d_in[0];
  const int*   ei   = (const int*)d_in[1];
  const float* ea   = (const float*)d_in[2];
  const float* onrm = (const float*)d_in[3];
  const float* Wq   = (const float*)d_in[4];  const float* bq    = (const float*)d_in[5];
  const float* Wk   = (const float*)d_in[6];  const float* bk    = (const float*)d_in[7];
  const float* Wv   = (const float*)d_in[8];  const float* bv    = (const float*)d_in[9];
  const float* We   = (const float*)d_in[10];
  const float* Wn   = (const float*)d_in[11]; const float* bn    = (const float*)d_in[12];
  const float* Wxy  = (const float*)d_in[13]; const float* bxy   = (const float*)d_in[14];
  const float* Wloc = (const float*)d_in[15]; const float* bloc  = (const float*)d_in[16];
  const float* Wobj = (const float*)d_in[17]; const float* bobj  = (const float*)d_in[18];
  const float* Wfus = (const float*)d_in[19]; const float* bfus  = (const float*)d_in[20];
  const float* Wcls = (const float*)d_in[21]; const float* bcls  = (const float*)d_in[22];
  const float* Wsk  = (const float*)d_in[23]; const float* bsk   = (const float*)d_in[24];
  const float* vocab= (const float*)d_in[25];
  float* out = (float*)d_out;

  // workspace carve-up
  char* cur = (char*)d_ws;
  auto alloc = [&](size_t b) -> char* {
    char* p = cur; cur += (b + 255) & ~(size_t)255; return p;
  };
  _Float16* xh     = (_Float16*)alloc((size_t)NN * KP * 2);
  _Float16* nrmh   = (_Float16*)alloc((size_t)NN * KP * 2);
  _Float16* Wqh    = (_Float16*)alloc((size_t)PACKW * 2);
  _Float16* Wkh    = (_Float16*)alloc((size_t)PACKW * 2);
  _Float16* Wvh    = (_Float16*)alloc((size_t)PACKW * 2);
  _Float16* Wskh   = (_Float16*)alloc((size_t)PACKW * 2);
  _Float16* WobjDh = (_Float16*)alloc((size_t)PACKW * 2);  // WobjA - WobjB
  _Float16* WobjBh = (_Float16*)alloc((size_t)PACKW * 2);
  _Float16* WlocMh = (_Float16*)alloc((size_t)PACKW * 2);  // Wloc rows 300..599
  _Float16* WlocSh = (_Float16*)alloc((size_t)PACKW * 2);  // Wloc rows 600..899
  float* qn    = (float*)alloc((size_t)NN * DD * 4);
  float* kn    = (float*)alloc((size_t)NN * DD * 4);
  float* vn    = (float*)alloc((size_t)NN * DD * 4);
  float* PD    = (float*)alloc((size_t)NN * DD * 4);
  float* PS    = (float*)alloc((size_t)NN * DD * 4);
  float* alpha = (float*)alloc((size_t)EE * HH * 4);
  float* eprob = (float*)alloc((size_t)EE * 8 * 4);
  unsigned* menc = (unsigned*)alloc((size_t)NN * HH * 4);
  float*    den  = (float*)alloc((size_t)NN * HH * 4);
  float* out7   = (float*)alloc(7 * DD * 4);
  float* W2buf  = (float*)alloc(2 * DD * 4);
  float* bias_s = (float*)alloc(DD * 4);
  float* vae    = (float*)alloc(DD * 4);
  float* vbe    = (float*)alloc(DD * 4);
  float* V5e    = (float*)alloc(5 * DD * 4);
  float* bfe    = (float*)alloc(DD * 4);

  const int TB = 256;
  const int wgrid = (PACKW + TB - 1) / TB;

  // stage f16 operands (weights packed into B-fragment order)
  k_conv_x<<<(NN * KP) / TB, TB, 0, stream>>>(x, xh);
  k_normal_h<<<(NN * KP) / TB, TB, 0, stream>>>(onrm, Wn, bn, nrmh);
  k_pack_w<<<wgrid, TB, 0, stream>>>(Wq, Wqh, DD, DD, DD);
  k_pack_w<<<wgrid, TB, 0, stream>>>(Wk, Wkh, DD, DD, DD);
  k_pack_w<<<wgrid, TB, 0, stream>>>(Wv, Wvh, DD, DD, DD);
  k_pack_w<<<wgrid, TB, 0, stream>>>(Wsk, Wskh, DD, DD, DD);
  k_pack_wdiff<<<wgrid, TB, 0, stream>>>(Wobj, WobjDh);
  k_pack_w<<<wgrid, TB, 0, stream>>>(Wobj + (size_t)DD * DD, WobjBh, DD, DD, DD);
  k_pack_w<<<wgrid, TB, 0, stream>>>(Wloc + (size_t)DD * DD, WlocMh, DD, DD, DD);
  k_pack_w<<<wgrid, TB, 0, stream>>>(Wloc + (size_t)2 * DD * DD, WlocSh, DD, DD, DD);

  // constant folds
  k_vocab_fus<<<(7 * DD + TB - 1) / TB, TB, 0, stream>>>(vocab, Wfus, out7);
  k_w2_bias<<<(3 * DD + TB - 1) / TB, TB, 0, stream>>>(Wxy, Wloc, bxy, bobj, bloc, W2buf, bias_s);
  k_we_fold<<<(8 * DD + TB - 1) / TB, TB, 0, stream>>>(out7, bfus, We, vae, vbe, V5e, bfe);

  // node-level WMMA GEMMs (M=50000, K=320 padded)
  dim3 gg(NN / 16, 3), gt(TB);
  gemm_wmma<<<gg, gt, 0, stream>>>(xh, Wqh, bq, nullptr, qn, DD, DD);
  gemm_wmma<<<gg, gt, 0, stream>>>(xh, Wkh, bk, nullptr, kn, DD, DD);
  gemm_wmma<<<gg, gt, 0, stream>>>(xh, Wvh, bv, nullptr, vn, DD, DD);
  gemm_wmma<<<gg, gt, 0, stream>>>(xh, Wskh, bsk, nullptr, out, DD, DD);  // root skip init
  gemm_wmma<<<gg, gt, 0, stream>>>(nrmh, WlocMh, bias_s, nullptr, PD, DD, DD);
  gemm_wmma<<<gg, gt, 0, stream>>>(xh, WobjDh, nullptr, PD, PD, DD, DD);
  gemm_wmma<<<gg, gt, 0, stream>>>(nrmh, WlocSh, nullptr, nullptr, PS, DD, DD);
  gemm_wmma<<<gg, gt, 0, stream>>>(xh, WobjBh, nullptr, PS, PS, DD, DD);

  // segment softmax state
  hipMemsetAsync(menc, 0, (size_t)NN * HH * 4, stream);  // key 0 == identity for max
  hipMemsetAsync(den, 0, (size_t)NN * HH * 4, stream);

  // edge passes (one wave32 per edge)
  edge_pass1<<<EE / 8, TB, 0, stream>>>(ei, ea, qn, kn, PD, PS, W2buf, Wcls, bcls,
                                        vae, vbe, V5e, bfe, alpha, eprob, menc);
  edge_pass2<<<(EE * HH + TB - 1) / TB, TB, 0, stream>>>(ei, alpha, menc, den);
  edge_pass3<<<EE / 8, TB, 0, stream>>>(ei, alpha, den, vn, eprob, vae, vbe, V5e, bfe, out);
}
